// HungarianMatcher_34196529610821
// MI455X (gfx1250) — matrix operand order repaired and checked by reading, not compiled, not run on hardware
//
#include <hip/hip_runtime.h>

typedef __attribute__((ext_vector_type(16))) _Float16 v16h;
typedef __attribute__((ext_vector_type(8)))  float    v8f;

namespace {
constexpr int kF = 36;                       // frames
constexpr int kO = 1000;                     // queries per frame (output rows)
constexpr int kT = 256;                      // target tracks (output cols)
constexpr int kC = 81;                       // classes
constexpr int kQ = kF * kO;                  // 36000
constexpr int kKReal = kF * kC;              // 2916 (GEMM K)
constexpr int kKTiles = (kKReal + 31) / 32;  // 92
constexpr int kOTiles = (kO + 15) / 16;      // 63
constexpr int kNTiles = kT / 16;             // 16
// packed fragment buffers: [tile][ktile][lane][16 halves]
constexpr size_t kAPackElems = (size_t)kOTiles * kKTiles * 32 * 16;  // 2,967,552
constexpr size_t kBPackElems = (size_t)kNTiles * kKTiles * 32 * 16;  //   753,664
}

// ---------------------------------------------------------------------------
// 1) softmax over 81 classes, store FP16 probabilities [F*O, C]
// ---------------------------------------------------------------------------
__global__ __launch_bounds__(256)
void softmax_kernel(const float* __restrict__ logits, _Float16* __restrict__ prob) {
  int q = blockIdx.x * blockDim.x + threadIdx.x;
  if (q >= kQ) return;
  const float* row = logits + (size_t)q * kC;
  float m = -3.402823466e38f;
  for (int c = 0; c < kC; ++c) m = fmaxf(m, row[c]);
  float s = 0.f;
  for (int c = 0; c < kC; ++c) s += __expf(row[c] - m);
  float inv = 1.f / s;
  _Float16* prow = prob + (size_t)q * kC;
  for (int c = 0; c < kC; ++c) prow[c] = (_Float16)(__expf(row[c] - m) * inv);
}

// ---------------------------------------------------------------------------
// 2a) pack A into per-lane WMMA fragment order (16-bit A 16x32 wave32 layout):
//     lane L (m = L&15, hi = L>>4), element e -> K = e + (e>=8 ? 8 : 0) + hi*8
// ---------------------------------------------------------------------------
__global__ __launch_bounds__(256)
void pack_a_kernel(const _Float16* __restrict__ P, _Float16* __restrict__ Ap) {
  size_t i = (size_t)blockIdx.x * blockDim.x + threadIdx.x;
  if (i >= kAPackElems) return;
  int e     = (int)(i & 15);
  int lane  = (int)((i >> 4) & 31);
  int tile  = (int)(i >> 9);
  int kt    = tile % kKTiles;
  int oTile = tile / kKTiles;
  int mrow = lane & 15, hi = lane >> 4;
  int kA = e + ((e >= 8) ? 8 : 0) + (hi << 3);
  int o = oTile * 16 + mrow;
  int k = kt * 32 + kA;
  _Float16 v = (_Float16)0.f;
  if (o < kO && k < kKReal) {
    int f = k / kC;
    int c = k - f * kC;
    v = P[((size_t)f * kO + o) * kC + c];    // out_prob_split[o, f, c]
  }
  Ap[i] = v;
}

// ---------------------------------------------------------------------------
// 2b) pack one-hot B directly from tgt_ids into fragment order
//     (16-bit B 32x16 wave32 layout): lane L (n = L&15, hi = L>>4),
//     element e -> K = e + (e>=8 ? 8 : 0) + hi*8
// ---------------------------------------------------------------------------
__global__ __launch_bounds__(256)
void pack_b_kernel(const int* __restrict__ tgt_ids, _Float16* __restrict__ Bp) {
  size_t i = (size_t)blockIdx.x * blockDim.x + threadIdx.x;
  if (i >= kBPackElems) return;
  int e     = (int)(i & 15);
  int lane  = (int)((i >> 4) & 31);
  int tile  = (int)(i >> 9);
  int kt    = tile % kKTiles;
  int nTile = tile / kKTiles;
  int ncol = lane & 15, hi = lane >> 4;
  int kB = e + ((e >= 8) ? 8 : 0) + (hi << 3);
  int t = nTile * 16 + ncol;
  int k = kt * 32 + kB;
  float v = 0.f;
  if (k < kKReal) {
    int f = k / kC;
    int c = k - f * kC;
    v = (tgt_ids[t * kF + f] == c) ? 1.f : 0.f;
  }
  Bp[i] = (_Float16)v;
}

// ---------------------------------------------------------------------------
// 3) class-cost GEMM: one wave computes a 16x64 slab (4 N-tiles reuse one A
//    fragment -> 4 v_wmma per A load). Fragments are pre-packed in per-lane
//    order: contiguous 32B vector loads, no LDS. Software-pipelined one tile
//    deep so k+1 fragment loads overlap the 4-WMMA burst on tile k.
// ---------------------------------------------------------------------------
__global__ __launch_bounds__(32)
void class_cost_wmma_kernel(const v16h* __restrict__ Ap,
                            const v16h* __restrict__ Bp,
                            float* __restrict__ out) {   // [kO, kT]
  const int lane  = threadIdx.x;       // 0..31 (wave32)
  const int slab  = blockIdx.x;        // 0..3 : 64-column slab
  const int oTile = blockIdx.y;        // 0..62

  const v16h* a_base  = Ap + ((size_t)oTile * kKTiles) * 32 + lane;
  const v16h* b_base0 = Bp + ((size_t)(slab * 4 + 0) * kKTiles) * 32 + lane;
  const v16h* b_base1 = Bp + ((size_t)(slab * 4 + 1) * kKTiles) * 32 + lane;
  const v16h* b_base2 = Bp + ((size_t)(slab * 4 + 2) * kKTiles) * 32 + lane;
  const v16h* b_base3 = Bp + ((size_t)(slab * 4 + 3) * kKTiles) * 32 + lane;

  v8f acc0 = {}, acc1 = {}, acc2 = {}, acc3 = {};

  // prologue: fragments for k-tile 0
  v16h a  = a_base[0];
  v16h b0 = b_base0[0];
  v16h b1 = b_base1[0];
  v16h b2 = b_base2[0];
  v16h b3 = b_base3[0];

  for (int kt = 1; kt < kKTiles; ++kt) {
    // issue next tile's loads before consuming current fragments
    v16h an  = a_base[(size_t)kt * 32];
    v16h bn0 = b_base0[(size_t)kt * 32];
    v16h bn1 = b_base1[(size_t)kt * 32];
    v16h bn2 = b_base2[(size_t)kt * 32];
    v16h bn3 = b_base3[(size_t)kt * 32];
    acc0 = __builtin_amdgcn_wmma_f32_16x16x32_f16(false, a, false, b0, (short)0, acc0, false, false);
    acc1 = __builtin_amdgcn_wmma_f32_16x16x32_f16(false, a, false, b1, (short)0, acc1, false, false);
    acc2 = __builtin_amdgcn_wmma_f32_16x16x32_f16(false, a, false, b2, (short)0, acc2, false, false);
    acc3 = __builtin_amdgcn_wmma_f32_16x16x32_f16(false, a, false, b3, (short)0, acc3, false, false);
    a = an; b0 = bn0; b1 = bn1; b2 = bn2; b3 = bn3;
  }
  // epilogue: last tile
  acc0 = __builtin_amdgcn_wmma_f32_16x16x32_f16(false, a, false, b0, (short)0, acc0, false, false);
  acc1 = __builtin_amdgcn_wmma_f32_16x16x32_f16(false, a, false, b1, (short)0, acc1, false, false);
  acc2 = __builtin_amdgcn_wmma_f32_16x16x32_f16(false, a, false, b2, (short)0, acc2, false, false);
  acc3 = __builtin_amdgcn_wmma_f32_16x16x32_f16(false, a, false, b3, (short)0, acc3, false, false);

  // D layout: VGPR r, lanes 0-15 -> M=r, lanes 16-31 -> M=r+8; N = lane&15
  const int mbase = (lane >> 4) << 3;
  const int ncol  = lane & 15;
  #pragma unroll
  for (int r = 0; r < 8; ++r) {
    int o = oTile * 16 + mbase + r;
    if (o < kO) {
      float* orow = out + (size_t)o * kT + slab * 64 + ncol;
      orow[0]  = acc0[r] * (-1.f / kF);
      orow[16] = acc1[r] * (-1.f / kF);
      orow[32] = acc2[r] * (-1.f / kF);
      orow[48] = acc3[r] * (-1.f / kF);
    }
  }
}

// ---------------------------------------------------------------------------
// 4) L1 bbox cost + GIoU cost, accumulated into out (runs after class kernel)
// ---------------------------------------------------------------------------
__global__ __launch_bounds__(256)
void box_cost_kernel(const float* __restrict__ pred_boxes,   // [F*O, 4] cxcywh
                     const float* __restrict__ tgt_bbox,     // [T*F, 4] cxcywh
                     float* __restrict__ out) {              // [kO, kT]
  int i = blockIdx.x * blockDim.x + threadIdx.x;
  if (i >= kO * kT) return;
  int o = i >> 8;
  int t = i & (kT - 1);
  const float4* pb = reinterpret_cast<const float4*>(pred_boxes);
  const float4* tb = reinterpret_cast<const float4*>(tgt_bbox);
  float l1 = 0.f, g = 0.f;
  for (int f = 0; f < kF; ++f) {
    float4 p = pb[f * kO + o];
    float4 q = tb[t * kF + f];
    l1 += fabsf(p.x - q.x) + fabsf(p.y - q.y) + fabsf(p.z - q.z) + fabsf(p.w - q.w);
    float ax0 = p.x - 0.5f * p.z, ay0 = p.y - 0.5f * p.w;
    float ax1 = p.x + 0.5f * p.z, ay1 = p.y + 0.5f * p.w;
    float bx0 = q.x - 0.5f * q.z, by0 = q.y - 0.5f * q.w;
    float bx1 = q.x + 0.5f * q.z, by1 = q.y + 0.5f * q.w;
    float iw = fmaxf(fminf(ax1, bx1) - fmaxf(ax0, bx0), 0.f);
    float ih = fmaxf(fminf(ay1, by1) - fmaxf(ay0, by0), 0.f);
    float inter = iw * ih;
    float areaA = (ax1 - ax0) * (ay1 - ay0);
    float areaB = (bx1 - bx0) * (by1 - by0);
    float uni = areaA + areaB - inter;
    float iou = inter / uni;
    float cw = fmaxf(fmaxf(ax1, bx1) - fminf(ax0, bx0), 0.f);
    float ch = fmaxf(fmaxf(ay1, by1) - fminf(ay0, by0), 0.f);
    float areaC = cw * ch;
    g += iou - (areaC - uni) / areaC;
  }
  out[i] += l1 * (1.f / (4.f * kF)) - g * (1.f / kF);
}

// ---------------------------------------------------------------------------
extern "C" void kernel_launch(void* const* d_in, const int* in_sizes, int n_in,
                              void* d_out, int out_size, void* d_ws, size_t ws_size,
                              hipStream_t stream) {
  (void)in_sizes; (void)n_in; (void)out_size; (void)ws_size;
  const float* pred_logits = (const float*)d_in[0];
  const float* pred_boxes  = (const float*)d_in[1];
  const float* tgt_bbox    = (const float*)d_in[2];
  const int*   tgt_ids     = (const int*)d_in[3];
  // d_in[4] = num_frames scalar (compile-time constant here)
  float* out = (float*)d_out;

  _Float16* P  = (_Float16*)d_ws;                    // 36000*81 f16  = 5,832,000 B
  _Float16* Ap = P + (size_t)kQ * kC;                // fragment-packed A, 5,935,104 B
  _Float16* Bp = Ap + kAPackElems;                   // fragment-packed B, 1,507,328 B
  // all offsets are multiples of 32 bytes -> v16h loads are aligned

  softmax_kernel<<<(kQ + 255) / 256, 256, 0, stream>>>(pred_logits, P);
  pack_a_kernel<<<(int)((kAPackElems + 255) / 256), 256, 0, stream>>>(P, Ap);
  pack_b_kernel<<<(int)((kBPackElems + 255) / 256), 256, 0, stream>>>(tgt_ids, Bp);
  class_cost_wmma_kernel<<<dim3(4, kOTiles), 32, 0, stream>>>(
      (const v16h*)Ap, (const v16h*)Bp, out);
  box_cost_kernel<<<(kO * kT + 255) / 256, 256, 0, stream>>>(pred_boxes, tgt_bbox, out);
}
